// CustomTransformerEncoderLayer_83391085019395
// MI455X (gfx1250) — compile-verified, matrix operand order checked by reading, not run on hardware
//
#include <hip/hip_runtime.h>
#include <hip/hip_bf16.h>
#include <stdint.h>

typedef __attribute__((ext_vector_type(16))) _Float16 v16h;
typedef __attribute__((ext_vector_type(8)))  float    v8f;

union AccU  { v8f  v; float    f[8]; };
union FragU { v16h v; uint32_t u[8]; };
union U32H  { uint32_t u; _Float16 h[2]; };

// Build a 16x32 f16 WMMA A/B fragment from one LDS row (row already selected
// per-lane).  ISA layout: VGPR v<4 holds K=(2v,2v+1)(+8 for lanes 16..31),
// VGPR v>=4 holds K=16+2(v-4)(+8 for hi lanes).
__device__ inline v16h frag_row(const _Float16* rp, int hi) {
  FragU f;
#pragma unroll
  for (int q = 0; q < 8; ++q) {
    const int kb = ((q & 3) * 2) + ((q >> 2) * 16) + (hi << 3);
    f.u[q] = *(const uint32_t*)(rp + kb);
  }
  return f.v;
}

__device__ inline v8f wmma32(v16h a, v16h b, v8f c) {
  return __builtin_amdgcn_wmma_f32_16x16x32_f16(false, a, false, b, (short)0, c,
                                                false, false);
}

// ---------------------------------------------------------------------------
// Generic f16 WMMA GEMM: C[M,N] = A[M,K] * B[K,N] (+bias[n]) (+addend[m,n])
// 8 waves / 256 threads.  Per-head batching via blockIdx.z with strides.
// ---------------------------------------------------------------------------
template <int BM, int BN, int MS, int NS>
__global__ void __launch_bounds__(256)
gemm_wmma(const _Float16* __restrict__ A, const _Float16* __restrict__ Bw,
          const float* __restrict__ bias, const float* __restrict__ addend,
          float* __restrict__ Cf, _Float16* __restrict__ Ch,
          int M, int N, int Kd, int lda, int ldb, int ldc,
          long long hsB, long long hsC, long long hsBias, int relu) {
  constexpr int WMW = BM / (MS * 16);
  constexpr int WNW = BN / (NS * 16);
  static_assert(WMW * WNW == 8, "need 8 waves");

  __shared__ _Float16 Al[BM][40];   // [m][k]  (k-tile of 32, +pad)
  __shared__ _Float16 Bl[BN][40];   // [n][k]  (transposed so k-pairs contiguous)

  const int tid  = threadIdx.x;
  const int lane = tid & 31;
  const int wv   = tid >> 5;
  const int hi   = lane >> 4;
  const int ll   = lane & 15;
  const int wm   = wv % WMW;
  const int wn   = wv / WMW;
  const int row0 = blockIdx.y * BM;
  const int col0 = blockIdx.x * BN;
  const long long head = blockIdx.z;
  const _Float16* Bh = Bw + head * hsB;

  const v8f zero = {0.f, 0.f, 0.f, 0.f, 0.f, 0.f, 0.f, 0.f};
  AccU acc[MS][NS];
#pragma unroll
  for (int i = 0; i < MS; ++i)
#pragma unroll
    for (int j = 0; j < NS; ++j) acc[i][j].v = zero;

  for (int kk = 0; kk < Kd; kk += 32) {
    __syncthreads();
    // Stage A tile (coalesced 32-bit loads, 2 f16 each)
#pragma unroll
    for (int i = 0; i < (BM * 16) / 256; ++i) {
      const int idx = tid + i * 256;
      const int r = idx >> 4, c = (idx & 15) << 1;
      *(uint32_t*)&Al[r][c] =
          *(const uint32_t*)(A + (long long)(row0 + r) * lda + kk + c);
    }
    // Stage B tile transposed: Bl[n][k] = B[kk+k][col0+n]
#pragma unroll
    for (int i = 0; i < (BN * 16) / 256; ++i) {
      const int idx = tid + i * 256;
      const int kg = idx / (BN / 2);
      const int np = (idx % (BN / 2)) << 1;
      U32H t;
      t.u = *(const uint32_t*)(Bh + (long long)(kk + kg) * ldb + col0 + np);
      Bl[np][kg]     = t.h[0];
      Bl[np + 1][kg] = t.h[1];
    }
    __syncthreads();

    v16h af[MS], bf[NS];
#pragma unroll
    for (int ms = 0; ms < MS; ++ms)
      af[ms] = frag_row(&Al[wm * MS * 16 + ms * 16 + ll][0], hi);
#pragma unroll
    for (int ns = 0; ns < NS; ++ns)
      bf[ns] = frag_row(&Bl[wn * NS * 16 + ns * 16 + ll][0], hi);
#pragma unroll
    for (int ms = 0; ms < MS; ++ms)
#pragma unroll
      for (int ns = 0; ns < NS; ++ns)
        acc[ms][ns].v = wmma32(af[ms], bf[ns], acc[ms][ns].v);
  }

  // Epilogue: C layout = lane holds col (lane%16), rows r (+8 for hi lanes).
#pragma unroll
  for (int ms = 0; ms < MS; ++ms) {
    const int rbase = row0 + wm * MS * 16 + ms * 16 + hi * 8;
#pragma unroll
    for (int ns = 0; ns < NS; ++ns) {
      const int cN = col0 + wn * NS * 16 + ns * 16 + ll;
      const float bvv = bias ? bias[head * hsBias + cN] : 0.f;
#pragma unroll
      for (int r = 0; r < 8; ++r) {
        const int rr = rbase + r;
        float val = acc[ms][ns].f[r] + bvv;
        if (relu) val = fmaxf(val, 0.f);
        if (addend) val += addend[(long long)rr * ldc + cN];
        const long long off = head * hsC + (long long)rr * ldc + cN;
        if (Cf) Cf[off] = val;
        if (Ch) Ch[off] = (_Float16)val;
      }
    }
  }
}

// ---------------------------------------------------------------------------
// Fused flash attention: one block = one (b,h) x 64-query tile, 4 waves.
// Each wave owns a 16-query sub-tile; key loop in blocks of 32 with online
// softmax (mask==0 -> -9e15), final scale[h]/l normalization.
// Q,K,V layout: [H][B*S][64] f16.  ctx: [B*S][1024] f16 (col = h*64+dk).
// ---------------------------------------------------------------------------
__global__ void __launch_bounds__(128)
attn_flash(const _Float16* __restrict__ Q, const _Float16* __restrict__ K,
           const _Float16* __restrict__ V, const int* __restrict__ mask,
           const float* __restrict__ scaleH, _Float16* __restrict__ ctx) {
  const int S = 1024, DK = 64, DM = 1024;
  __shared__ _Float16 Ql[64][72];      // [q][dk]
  __shared__ _Float16 Kl[32][72];      // [key][dk]  (k-pairs contiguous for QK^T)
  __shared__ _Float16 Vt[64][40];      // [dk][key]  (k-pairs contiguous for P*V)
  __shared__ _Float16 Pl[4][16][40];   // per-wave P relayout scratch

  const int tid  = threadIdx.x;
  const int lane = tid & 31;
  const int wv   = tid >> 5;
  const int hi   = lane >> 4;
  const int ll   = lane & 15;
  const int h  = blockIdx.y;
  const int b  = blockIdx.z;
  const int q0 = blockIdx.x * 64;
  const long long mrow = (long long)b * S + q0;
  const _Float16* Qb = Q + ((long long)h * 4096 + mrow) * DK;

#pragma unroll
  for (int i = 0; i < 16; ++i) {
    const int idx = tid + i * 128;
    const int r = idx >> 5, c = (idx & 31) << 1;
    *(uint32_t*)&Ql[r][c] = *(const uint32_t*)(Qb + r * DK + c);
  }
  __syncthreads();

  const v16h qa0 = frag_row(&Ql[wv * 16 + ll][0], hi);
  const v16h qa1 = frag_row(&Ql[wv * 16 + ll][32], hi);

  const v8f zero = {0.f, 0.f, 0.f, 0.f, 0.f, 0.f, 0.f, 0.f};
  float mrun[8], lrun[8];
  AccU o[4];
#pragma unroll
  for (int r = 0; r < 8; ++r) { mrun[r] = -3e38f; lrun[r] = 0.f; }
#pragma unroll
  for (int d = 0; d < 4; ++d) o[d].v = zero;

  const float inv = 0.125f;  // 1/sqrt(64)

  for (int kb = 0; kb < S; kb += 32) {
    __syncthreads();
    const _Float16* Kb = K + ((long long)h * 4096 + (long long)b * S + kb) * DK;
    const _Float16* Vb = V + ((long long)h * 4096 + (long long)b * S + kb) * DK;
#pragma unroll
    for (int i = 0; i < 8; ++i) {
      const int idx = tid + i * 128;
      const int r = idx >> 5, c = (idx & 31) << 1;
      *(uint32_t*)&Kl[r][c] = *(const uint32_t*)(Kb + r * DK + c);
      U32H t;
      t.u = *(const uint32_t*)(Vb + r * DK + c);
      Vt[c][r]     = t.h[0];
      Vt[c + 1][r] = t.h[1];
    }
    __syncthreads();

    float p0[8], p1[8];
    {  // keys kb .. kb+15
      const v16h kf0 = frag_row(&Kl[ll][0], hi);
      const v16h kf1 = frag_row(&Kl[ll][32], hi);
      AccU s; s.v = zero;
      s.v = wmma32(qa0, kf0, s.v);
      s.v = wmma32(qa1, kf1, s.v);
      const int mk = mask[(long long)b * S + kb + ll];
#pragma unroll
      for (int r = 0; r < 8; ++r) p0[r] = mk ? s.f[r] * inv : -9e15f;
    }
    {  // keys kb+16 .. kb+31
      const v16h kf0 = frag_row(&Kl[16 + ll][0], hi);
      const v16h kf1 = frag_row(&Kl[16 + ll][32], hi);
      AccU s; s.v = zero;
      s.v = wmma32(qa0, kf0, s.v);
      s.v = wmma32(qa1, kf1, s.v);
      const int mk = mask[(long long)b * S + kb + 16 + ll];
#pragma unroll
      for (int r = 0; r < 8; ++r) p1[r] = mk ? s.f[r] * inv : -9e15f;
    }

    // Online softmax per row; rows of a tile live across a 16-lane half.
#pragma unroll
    for (int r = 0; r < 8; ++r) {
      float mloc = fmaxf(p0[r], p1[r]);
      for (int xm = 1; xm < 16; xm <<= 1)
        mloc = fmaxf(mloc, __shfl_xor(mloc, xm, 32));
      const float mnew = fmaxf(mrun[r], mloc);
      const float corr = __expf(mrun[r] - mnew);
      mrun[r] = mnew;
      p0[r] = __expf(p0[r] - mnew);
      p1[r] = __expf(p1[r] - mnew);
      float rs = p0[r] + p1[r];
      for (int xm = 1; xm < 16; xm <<= 1) rs += __shfl_xor(rs, xm, 32);
      lrun[r] = lrun[r] * corr + rs;
#pragma unroll
      for (int d = 0; d < 4; ++d) o[d].f[r] *= corr;
    }

    // Relayout P (C-layout f32) -> A-fragment (f16) via per-wave LDS scratch.
#pragma unroll
    for (int r = 0; r < 8; ++r) {
      Pl[wv][r + hi * 8][ll]      = (_Float16)p0[r];
      Pl[wv][r + hi * 8][16 + ll] = (_Float16)p1[r];
    }
    __syncthreads();
    const v16h pa = frag_row(&Pl[wv][ll][0], hi);
#pragma unroll
    for (int d = 0; d < 4; ++d) {
      const v16h vb = frag_row(&Vt[d * 16 + ll][0], hi);
      o[d].v = wmma32(pa, vb, o[d].v);
    }
  }

  const float sc = scaleH[h];
#pragma unroll
  for (int r = 0; r < 8; ++r) {
    const float fct = sc / lrun[r];
    const long long grow = mrow + wv * 16 + r + hi * 8;
#pragma unroll
    for (int d = 0; d < 4; ++d)
      ctx[grow * DM + h * 64 + d * 16 + ll] = (_Float16)(o[d].f[r] * fct);
  }
}

// ---------------------------------------------------------------------------
// LayerNorm over D=1024, one block per row, f16 output for WMMA GEMMs.
// ---------------------------------------------------------------------------
__global__ void __launch_bounds__(256)
ln_f16(const float* __restrict__ X, const float* __restrict__ g,
       const float* __restrict__ bb, _Float16* __restrict__ Y) {
  const int D = 1024;
  __shared__ float s1[256], s2[256];
  const int row = blockIdx.x, tid = threadIdx.x;
  const float4 x4 = *(const float4*)(X + (long long)row * D + tid * 4);
  float sum = x4.x + x4.y + x4.z + x4.w;
  float sq  = x4.x * x4.x + x4.y * x4.y + x4.z * x4.z + x4.w * x4.w;
  s1[tid] = sum; s2[tid] = sq;
  __syncthreads();
  for (int off = 128; off > 0; off >>= 1) {
    if (tid < off) { s1[tid] += s1[tid + off]; s2[tid] += s2[tid + off]; }
    __syncthreads();
  }
  const float m = s1[0] * (1.f / D);
  const float var = s2[0] * (1.f / D) - m * m;
  const float rstd = rsqrtf(var + 1e-5f);
#pragma unroll
  for (int j = 0; j < 4; ++j) {
    const int c = tid * 4 + j;
    const float xv = (&x4.x)[j];
    Y[(long long)row * D + c] = (_Float16)((xv - m) * rstd * g[c] + bb[c]);
  }
}

__global__ void __launch_bounds__(256)
add_f32(const float* __restrict__ a, const float* __restrict__ b,
        float* __restrict__ c, int n) {
  const int i = blockIdx.x * 256 + threadIdx.x;
  if (i < n) c[i] = a[i] + b[i];
}

__global__ void __launch_bounds__(256)
f2h(const float* __restrict__ a, _Float16* __restrict__ b, int n) {
  const int i = blockIdx.x * 256 + threadIdx.x;
  if (i < n) b[i] = (_Float16)a[i];
}

// ---------------------------------------------------------------------------
extern "C" void kernel_launch(void* const* d_in, const int* in_sizes, int n_in,
                              void* d_out, int out_size, void* d_ws,
                              size_t ws_size, hipStream_t stream) {
  (void)in_sizes; (void)n_in; (void)out_size; (void)ws_size;
  const float* src  = (const float*)d_in[0];
  const int*   mask = (const int*)d_in[1];
  const float* Wq   = (const float*)d_in[2];
  const float* bq   = (const float*)d_in[3];
  const float* Wk   = (const float*)d_in[4];
  const float* bk   = (const float*)d_in[5];
  const float* Wv   = (const float*)d_in[6];
  const float* bv   = (const float*)d_in[7];
  const float* scl  = (const float*)d_in[8];
  const float* Wo   = (const float*)d_in[9];
  const float* bo   = (const float*)d_in[10];
  const float* ln1g = (const float*)d_in[11];
  const float* ln1b = (const float*)d_in[12];
  const float* ln2g = (const float*)d_in[13];
  const float* ln2b = (const float*)d_in[14];
  const float* W1   = (const float*)d_in[15];
  const float* b1   = (const float*)d_in[16];
  const float* W2   = (const float*)d_in[17];
  const float* b2   = (const float*)d_in[18];

  const int M = 4096, DM = 1024, DFF = 4096, H = 16;

  char* p = (char*)d_ws;
  auto carve = [&](size_t bytes) -> void* {
    void* r = (void*)p;
    p += (bytes + 255) & ~(size_t)255;
    return r;
  };
  _Float16* xf16  = (_Float16*)carve((size_t)M * DM * 2);
  _Float16* wq16  = (_Float16*)carve((size_t)DM * DM * 2);
  _Float16* wk16  = (_Float16*)carve((size_t)DM * DM * 2);
  _Float16* wv16  = (_Float16*)carve((size_t)DM * DM * 2);
  _Float16* wo16  = (_Float16*)carve((size_t)DM * DM * 2);
  _Float16* w116  = (_Float16*)carve((size_t)DM * DFF * 2);
  _Float16* w216  = (_Float16*)carve((size_t)DFF * DM * 2);
  _Float16* k16   = (_Float16*)carve((size_t)H * M * 64 * 2);
  _Float16* v16   = (_Float16*)carve((size_t)H * M * 64 * 2);
  _Float16* q16   = (_Float16*)carve((size_t)H * M * 64 * 2);
  _Float16* ctx16 = (_Float16*)carve((size_t)M * DM * 2);
  _Float16* out16 = (_Float16*)carve((size_t)M * DM * 2);
  _Float16* y16   = (_Float16*)carve((size_t)M * DM * 2);
  _Float16* h16   = (_Float16*)carve((size_t)M * DFF * 2);
  float*    out32 = (float*)carve((size_t)M * DM * 4);
  float*    src2  = (float*)carve((size_t)M * DM * 4);

  // Weight f32 -> f16 conversion (layout-preserving).
  f2h<<<(DM * DM + 255) / 256, 256, 0, stream>>>(Wq, wq16, DM * DM);
  f2h<<<(DM * DM + 255) / 256, 256, 0, stream>>>(Wk, wk16, DM * DM);
  f2h<<<(DM * DM + 255) / 256, 256, 0, stream>>>(Wv, wv16, DM * DM);
  f2h<<<(DM * DM + 255) / 256, 256, 0, stream>>>(Wo, wo16, DM * DM);
  f2h<<<(DM * DFF + 255) / 256, 256, 0, stream>>>(W1, w116, DM * DFF);
  f2h<<<(DFF * DM + 255) / 256, 256, 0, stream>>>(W2, w216, DFF * DM);

  // x = LN1(src)
  ln_f16<<<M, 256, 0, stream>>>(src, ln1g, ln1b, xf16);

  // K/V projections, per head (Wk[h] is [1024x64] with ldb=64).
  gemm_wmma<128, 64, 4, 1><<<dim3(1, M / 128, H), 256, 0, stream>>>(
      xf16, wk16, bk, nullptr, nullptr, k16, M, 64, DM, DM, 64, 64,
      (long long)DM * 64, (long long)M * 64, 64, 0);
  gemm_wmma<128, 64, 4, 1><<<dim3(1, M / 128, H), 256, 0, stream>>>(
      xf16, wv16, bv, nullptr, nullptr, v16, M, 64, DM, DM, 64, 64,
      (long long)DM * 64, (long long)M * 64, 64, 0);

  const _Float16* cur = xf16;
  for (int it = 0; it < 3; ++it) {
    // Q projection per head from current stream.
    gemm_wmma<128, 64, 4, 1><<<dim3(1, M / 128, H), 256, 0, stream>>>(
        cur, wq16, bq, nullptr, nullptr, q16, M, 64, DM, DM, 64, 64,
        (long long)DM * 64, (long long)M * 64, 64, 0);
    // Fused attention -> ctx16 [M][1024].
    attn_flash<<<dim3(16, H, 4), 128, 0, stream>>>(q16, k16, v16, mask, scl,
                                                   ctx16);
    // out = ctx @ Wo + bo  (f32 + f16 copy for next iteration's Q proj).
    gemm_wmma<128, 128, 4, 2><<<dim3(DM / 128, M / 128, 1), 256, 0, stream>>>(
        ctx16, wo16, bo, nullptr, out32, out16, M, DM, DM, DM, DM, DM, 0, 0, 0,
        0);
    cur = out16;
  }

  // src2 = src + out ; y = LN2(src2)
  add_f32<<<(M * DM + 255) / 256, 256, 0, stream>>>(src, out32, src2, M * DM);
  ln_f16<<<M, 256, 0, stream>>>(src2, ln2g, ln2b, y16);

  // FFN: h = relu(y @ W1 + b1); out = h @ W2 + b2 + src2 -> d_out
  gemm_wmma<128, 128, 4, 2><<<dim3(DFF / 128, M / 128, 1), 256, 0, stream>>>(
      y16, w116, b1, nullptr, nullptr, h16, M, DFF, DM, DM, DFF, DFF, 0, 0, 0,
      1);
  gemm_wmma<128, 128, 4, 2><<<dim3(DM / 128, M / 128, 1), 256, 0, stream>>>(
      h16, w216, b2, src2, (float*)d_out, nullptr, M, DM, DFF, DFF, DM, DM, 0,
      0, 0, 0);
}
